// VectorQuantizer_28552942584036
// MI455X (gfx1250) — matrix operand order repaired and checked by reading, not compiled, not run on hardware
//
#include <hip/hip_runtime.h>
#include <hip/hip_bf16.h>

typedef __attribute__((ext_vector_type(16))) _Float16 v16h;
typedef __attribute__((ext_vector_type(8)))  float    v8f;

#define LATENT_DIM       64
#define CODEBOOK_SIZE    1024
#define ROW_TILES        4                                // 16-row tiles per wave
#define ROWS_PER_WAVE    (16 * ROW_TILES)                 // 64
#define WAVES_PER_WG     8
#define ROWS_PER_WG      (ROWS_PER_WAVE * WAVES_PER_WG)   // 512
#define N_TILES          (CODEBOOK_SIZE / 16)             // 64

// Dynamic LDS layout:
//   [0, 131072)      : codebook as f16, code-major (1024 codes x 64 halfs)
//   [131072, 135168) : |w|^2 per code (1024 floats)
#define LDS_CB_BYTES   (CODEBOOK_SIZE * LATENT_DIM * 2)
#define LDS_TOTAL      (LDS_CB_BYTES + CODEBOOK_SIZE * 4)

__global__ __launch_bounds__(256) void
vq_argmin_wmma_kernel(const float* __restrict__ x,
                      const float* __restrict__ codebook,
                      float* __restrict__ qout,   // [262144 * 64] fp32
                      int*   __restrict__ iout)   // [262144] int32
{
    extern __shared__ char smem[];
    _Float16* cb_h = (_Float16*)smem;                 // f16 codebook, code-major
    float*    w2s  = (float*)(smem + LDS_CB_BYTES);   // |w|^2

    const int tid  = threadIdx.x;
    const int lane = tid & 31;
    const int wave = tid >> 5;

    // ---- Stage 1: single pass over codebook: fp32 -> f16 into LDS + |w|^2 ----
    for (int c = tid; c < CODEBOOK_SIZE; c += 256) {
        const float4* p = (const float4*)(codebook + (size_t)c * LATENT_DIM);
        uint2* dst = (uint2*)(cb_h + (size_t)c * LATENT_DIM);
        float s = 0.f;
        #pragma unroll
        for (int j = 0; j < LATENT_DIM / 4; ++j) {
            float4 v = p[j];
            s += v.x * v.x + v.y * v.y + v.z * v.z + v.w * v.w;
            union { _Float16 h[4]; uint2 u; } pk;
            pk.h[0] = (_Float16)v.x; pk.h[1] = (_Float16)v.y;
            pk.h[2] = (_Float16)v.z; pk.h[3] = (_Float16)v.w;
            dst[j] = pk.u;
        }
        w2s[c] = s;
    }
    __syncthreads();

    // ---- Stage 2: build A fragments: 4 row-tiles (64 rows of x) per wave ----
    // A k order used here (16x32, matched by B): lanes 0-15 -> K 0..15; lanes 16-31 -> K 16..31
    const int rowBase = blockIdx.x * ROWS_PER_WG + wave * ROWS_PER_WAVE;
    const int m       = lane & 15;
    const int kbase   = (lane >= 16) ? 16 : 0;

    v16h a0[ROW_TILES], a1[ROW_TILES];
    #pragma unroll
    for (int rt = 0; rt < ROW_TILES; ++rt) {
        const float* xr = x + (size_t)(rowBase + rt * 16 + m) * LATENT_DIM;
        const float4* q  = (const float4*)(xr + kbase);
        const float4* r  = (const float4*)(xr + kbase + 8);
        const float4* q2 = (const float4*)(xr + 32 + kbase);
        const float4* r2 = (const float4*)(xr + 32 + kbase + 8);
        float4 v0 = q[0],  v1 = q[1],  v2 = r[0],  v3 = r[1];
        float4 v4 = q2[0], v5 = q2[1], v6 = r2[0], v7 = r2[1];
        v16h A0, A1;
        A0[0]=(_Float16)v0.x;  A0[1]=(_Float16)v0.y;  A0[2]=(_Float16)v0.z;  A0[3]=(_Float16)v0.w;
        A0[4]=(_Float16)v1.x;  A0[5]=(_Float16)v1.y;  A0[6]=(_Float16)v1.z;  A0[7]=(_Float16)v1.w;
        A0[8]=(_Float16)v2.x;  A0[9]=(_Float16)v2.y;  A0[10]=(_Float16)v2.z; A0[11]=(_Float16)v2.w;
        A0[12]=(_Float16)v3.x; A0[13]=(_Float16)v3.y; A0[14]=(_Float16)v3.z; A0[15]=(_Float16)v3.w;
        A1[0]=(_Float16)v4.x;  A1[1]=(_Float16)v4.y;  A1[2]=(_Float16)v4.z;  A1[3]=(_Float16)v4.w;
        A1[4]=(_Float16)v5.x;  A1[5]=(_Float16)v5.y;  A1[6]=(_Float16)v5.z;  A1[7]=(_Float16)v5.w;
        A1[8]=(_Float16)v6.x;  A1[9]=(_Float16)v6.y;  A1[10]=(_Float16)v6.z; A1[11]=(_Float16)v6.w;
        A1[12]=(_Float16)v7.x; A1[13]=(_Float16)v7.y; A1[14]=(_Float16)v7.z; A1[15]=(_Float16)v7.w;
        a0[rt] = A0; a1[rt] = A1;
    }

    // ---- Stage 3: sweep codebook tiles ----
    // C-operand preload: acc = x.w - |w|^2/2  (argmin distance == argmax acc).
    // B layout (32x16): lane -> column n = lane&15; lanes 0-15 carry K 0..15,
    // lanes 16-31 carry K 16..31 -> contiguous 32B slices of the code's f16 row.
    float best_val[ROW_TILES][8];
    int   best_idx[ROW_TILES][8];
    #pragma unroll
    for (int rt = 0; rt < ROW_TILES; ++rt)
        #pragma unroll
        for (int i = 0; i < 8; ++i) { best_val[rt][i] = -3.4e38f; best_idx[rt][i] = 0; }

    const int laneKOff = (lane >= 16) ? 16 : 0;

    for (int t = 0; t < N_TILES; ++t) {
        const int ncol = t * 16 + m;
        const float wn = w2s[ncol];                      // hoisted; hides behind WMMAs
        const v16h* bp = (const v16h*)(cb_h + (size_t)ncol * LATENT_DIM + laneKOff);
        v16h b0 = bp[0];          // K  0..31 slice for this lane half
        v16h b1 = bp[2];          // K 32..63 slice for this lane half

        const float c = -0.5f * wn;                      // exact (power-of-2 scale)
        v8f acc0;
        #pragma unroll
        for (int i = 0; i < 8; ++i) acc0[i] = c;         // shared C for all 4 row-tiles

        #pragma unroll
        for (int rt = 0; rt < ROW_TILES; ++rt) {
            v8f acc;
            acc = __builtin_amdgcn_wmma_f32_16x16x32_f16(false, a0[rt], false, b0,
                                                         (short)0, acc0, false, false);
            acc = __builtin_amdgcn_wmma_f32_16x16x32_f16(false, a1[rt], false, b1,
                                                         (short)0, acc, false, false);
            #pragma unroll
            for (int i = 0; i < 8; ++i) {
                // strict > keeps the earliest (smallest) index on ties
                if (acc[i] > best_val[rt][i]) { best_val[rt][i] = acc[i]; best_idx[rt][i] = ncol; }
            }
        }
    }

    // ---- Stage 4: max-reduce (val,idx) across the 16-lane column groups ----
    #pragma unroll
    for (int mask = 1; mask < 16; mask <<= 1) {
        #pragma unroll
        for (int rt = 0; rt < ROW_TILES; ++rt) {
            #pragma unroll
            for (int i = 0; i < 8; ++i) {
                float ov = __shfl_xor(best_val[rt][i], mask, 32);
                int   oi = __shfl_xor(best_idx[rt][i], mask, 32);
                if (ov > best_val[rt][i] ||
                    (ov == best_val[rt][i] && oi < best_idx[rt][i])) {
                    best_val[rt][i] = ov; best_idx[rt][i] = oi;
                }
            }
        }
    }
    // Lanes 0-15 hold rows {tile base + 0..7} (slot i -> row i); lanes 16-31 rows 8..15.

    // ---- Stage 5: gather fp32 codebook rows + emit indices (coalesced) ----
    #pragma unroll
    for (int rt = 0; rt < ROW_TILES; ++rt) {
        #pragma unroll
        for (int r = 0; r < 16; ++r) {
            int idx = __shfl(best_idx[rt][r & 7], (r < 8) ? 0 : 16, 32);
            size_t row = (size_t)rowBase + rt * 16 + r;
            const float2* src = (const float2*)(codebook + (size_t)idx * LATENT_DIM);
            ((float2*)(qout + row * LATENT_DIM))[lane] = src[lane];  // 256B/row, 32 lanes
            if (lane == 0) iout[row] = idx;
        }
    }
}

extern "C" void kernel_launch(void* const* d_in, const int* in_sizes, int n_in,
                              void* d_out, int out_size, void* d_ws, size_t ws_size,
                              hipStream_t stream) {
    const float* x        = (const float*)d_in[0];   // [16,128,128,64] fp32
    const float* codebook = (const float*)d_in[1];   // [1024,64] fp32

    const int n_rows = in_sizes[0] / LATENT_DIM;     // 262144

    float* qout = (float*)d_out;                                   // fp32 quantized
    int*   iout = (int*)(qout + (size_t)n_rows * LATENT_DIM);      // int32 indices

    const int blocks = n_rows / ROWS_PER_WG;         // 512
    vq_argmin_wmma_kernel<<<blocks, 256, LDS_TOTAL, stream>>>(x, codebook, qout, iout);
}
